// Net_31078383354359
// MI455X (gfx1250) — matrix operand order repaired and checked by reading, not compiled, not run on hardware
//
#include <hip/hip_runtime.h>

// ---------------------------------------------------------------------------
// 2-layer GCN for MI455X (gfx1250).
//   deg -> dinv -> [GEMM1 wmma f16, prepacked W frags] -> edge scatter
//   -> relu+bias+selfloop (f16) -> [GEMM2 wmma f16] -> edge scatter
//   -> bias+selfloop+log_softmax (wave/row, shfl reductions)
// ---------------------------------------------------------------------------

#define NNODES 100000
#define NEDGES 1600000
#define FIN    512
#define HID    64
#define NCLS   40

typedef __attribute__((ext_vector_type(16))) _Float16 v16h;
typedef __attribute__((ext_vector_type(8)))  _Float16 v8h;
typedef __attribute__((ext_vector_type(8)))  float    v8f;

__device__ __forceinline__ void atomAddF(float* p, float v) {
    unsafeAtomicAdd(p, v);   // lowers to global_atomic_add_f32
}

// ---------------- degree / normalization ----------------
__global__ void k_fill1(float* deg) {
    int i = blockIdx.x * blockDim.x + threadIdx.x;
    if (i < NNODES) deg[i] = 1.0f;                 // self-loop
}

__global__ void k_deg(const int* __restrict__ dst, float* __restrict__ deg) {
    int e = blockIdx.x * blockDim.x + threadIdx.x;
    if (e < NEDGES) atomAddF(&deg[dst[e]], 1.0f);
}

__global__ void k_rsqrt(float* deg) {
    int i = blockIdx.x * blockDim.x + threadIdx.x;
    if (i < NNODES) deg[i] = rsqrtf(deg[i]);       // deg >= 1 always
}

// ---------------- W prepack into WMMA B-fragment order (f16) ----------------
// Fragment (kt, nt): per lane, element e holds W[kt*32 + (lane>>4)*16 + e][nt*16 + (lane&15)]
// packed at wp[(kt*NT + nt)*512 + lane*16 + e]   (512 halves = one wave fragment)
__global__ void k_packW1(const float* __restrict__ W1, _Float16* __restrict__ wp) {
    const int wid = (blockIdx.x * blockDim.x + threadIdx.x) >> 5;
    if (wid >= (FIN / 32) * 4) return;             // 64 fragments
    const int lane = threadIdx.x & 31;
    const int kt = wid >> 2, nt = wid & 3;
    const int krow = kt * 32 + (lane >> 4) * 16;
    const int col  = nt * 16 + (lane & 15);
    _Float16* d = wp + (size_t)wid * 512 + lane * 16;
    #pragma unroll
    for (int e = 0; e < 16; ++e)
        d[e] = (_Float16)W1[(size_t)(krow + e) * HID + col];
}

__global__ void k_packW2(const float* __restrict__ W2, _Float16* __restrict__ wp) {
    const int wid = (blockIdx.x * blockDim.x + threadIdx.x) >> 5;
    if (wid >= (HID / 32) * 3) return;             // 6 fragments (40 cols padded to 48)
    const int lane = threadIdx.x & 31;
    const int kt = wid / 3, nt = wid % 3;
    const int krow = kt * 32 + (lane >> 4) * 16;
    const int col  = nt * 16 + (lane & 15);
    _Float16* d = wp + (size_t)wid * 512 + lane * 16;
    #pragma unroll
    for (int e = 0; e < 16; ++e)
        d[e] = (col < NCLS) ? (_Float16)W2[(size_t)(krow + e) * NCLS + col] : (_Float16)0.0f;
}

// ---------------- GEMM1: h = x @ W1  (f16 WMMA, f32 accum) ----------------
// One wave computes a 16(M) x 64(N) tile.  K loop: 512/32 = 16 steps.
__global__ void k_gemm1(const float* __restrict__ x,
                        const _Float16* __restrict__ w1p,
                        float* __restrict__ h) {
    const int wid = (blockIdx.x * blockDim.x + threadIdx.x) >> 5;
    if (wid >= NNODES / 16) return;                // wave-uniform guard (EXEC stays full)
    const int lane = threadIdx.x & 31;
    const int n0   = lane & 15;
    const int hf   = lane >> 4;
    const int m    = wid * 16 + n0;                // A row for this lane
    const float* xrow = x + (size_t)m * FIN;

    v8f acc0 = {}, acc1 = {}, acc2 = {}, acc3 = {};

    for (int kb = 0; kb < FIN; kb += 32) {
        // ---- A fragment: elems 0..15 -> K = kb+hf*16.. (contiguous)
        v16h a;
        {
            const float4* p0 = (const float4*)(xrow + kb + hf * 16);
            float4 u = p0[0], v = p0[1];
            a[0]=(_Float16)u.x; a[1]=(_Float16)u.y; a[2]=(_Float16)u.z; a[3]=(_Float16)u.w;
            a[4]=(_Float16)v.x; a[5]=(_Float16)v.y; a[6]=(_Float16)v.z; a[7]=(_Float16)v.w;
            const float4* p1 = (const float4*)(xrow + kb + 8 + hf * 16);
            u = p1[0]; v = p1[1];
            a[8] =(_Float16)u.x; a[9] =(_Float16)u.y; a[10]=(_Float16)u.z; a[11]=(_Float16)u.w;
            a[12]=(_Float16)v.x; a[13]=(_Float16)v.y; a[14]=(_Float16)v.z; a[15]=(_Float16)v.w;
        }
        // ---- B fragments: contiguous 32B per lane, L2-resident
        const _Float16* wk = w1p + (size_t)(kb >> 5) * 4 * 512 + lane * 16;
        v16h b0 = *(const v16h*)(wk);
        v16h b1 = *(const v16h*)(wk + 512);
        v16h b2 = *(const v16h*)(wk + 1024);
        v16h b3 = *(const v16h*)(wk + 1536);

        acc0 = __builtin_amdgcn_wmma_f32_16x16x32_f16(false, a, false, b0, (short)0, acc0, false, false);
        acc1 = __builtin_amdgcn_wmma_f32_16x16x32_f16(false, a, false, b1, (short)0, acc1, false, false);
        acc2 = __builtin_amdgcn_wmma_f32_16x16x32_f16(false, a, false, b2, (short)0, acc2, false, false);
        acc3 = __builtin_amdgcn_wmma_f32_16x16x32_f16(false, a, false, b3, (short)0, acc3, false, false);
    }

    // ---- store D: row (hf*8 + r), col tile*16 + n0
    float* hbase = h + (size_t)(wid * 16) * HID;
    #pragma unroll
    for (int r = 0; r < 8; ++r) {
        const size_t row = (size_t)(hf * 8 + r) * HID + n0;
        hbase[row +  0] = acc0[r];
        hbase[row + 16] = acc1[r];
        hbase[row + 32] = acc2[r];
        hbase[row + 48] = acc3[r];
    }
}

// ---------------- scatter layer 1: one wave per edge, 64 feats ----------------
__global__ void k_scatter1(const int* __restrict__ src, const int* __restrict__ dst,
                           const float* __restrict__ dinv,
                           const float* __restrict__ h, float* __restrict__ agg) {
    const int e = (blockIdx.x * blockDim.x + threadIdx.x) >> 5;
    if (e >= NEDGES) return;
    const int lane = threadIdx.x & 31;
    const int s = src[e], d = dst[e];
    const float nrm = dinv[s] * dinv[d];
    const float* hs = h   + (size_t)s * HID;
    float*       ad = agg + (size_t)d * HID;
    atomAddF(&ad[lane],      hs[lane]      * nrm);
    atomAddF(&ad[lane + 32], hs[lane + 32] * nrm);
}

// ---------------- finalize1: +selfloop, +bias, relu, (idx sign), -> f16 ----------------
__global__ void k_fin1(const float* __restrict__ agg, const float* __restrict__ h,
                       const float* __restrict__ dinv, const float* __restrict__ b1,
                       const int* __restrict__ idxf, _Float16* __restrict__ hrelu) {
    const int i = blockIdx.x * blockDim.x + threadIdx.x;
    if (i >= NNODES * HID) return;
    const int node = i >> 6;
    const int f    = i & 63;
    const float di = dinv[node];
    float v = agg[i] + di * di * h[i] + b1[f];
    v = v > 0.0f ? v : 0.0f;
    if (idxf[0] != 0) v = -v;
    hrelu[i] = (_Float16)v;
}

// ---------------- GEMM2: z = hrelu @ W2  (16 x 48-padded tile, K=64) ----------------
__global__ void k_gemm2(const _Float16* __restrict__ hr,
                        const _Float16* __restrict__ w2p,
                        float* __restrict__ z) {
    const int wid = (blockIdx.x * blockDim.x + threadIdx.x) >> 5;
    if (wid >= NNODES / 16) return;
    const int lane = threadIdx.x & 31;
    const int n0   = lane & 15;
    const int hf   = lane >> 4;
    const int m    = wid * 16 + n0;
    const _Float16* arow = hr + (size_t)m * HID;

    v8f acc0 = {}, acc1 = {}, acc2 = {};

    #pragma unroll
    for (int kb = 0; kb < HID; kb += 32) {
        v16h a;
        {
            v8h lo = *(const v8h*)(arow + kb + hf * 16);
            v8h hi = *(const v8h*)(arow + kb + 8 + hf * 16);
            #pragma unroll
            for (int e = 0; e < 8; ++e) { a[e] = lo[e]; a[8 + e] = hi[e]; }
        }
        const _Float16* wk = w2p + (size_t)(kb >> 5) * 3 * 512 + lane * 16;
        v16h b0 = *(const v16h*)(wk);
        v16h b1 = *(const v16h*)(wk + 512);
        v16h b2 = *(const v16h*)(wk + 1024);

        acc0 = __builtin_amdgcn_wmma_f32_16x16x32_f16(false, a, false, b0, (short)0, acc0, false, false);
        acc1 = __builtin_amdgcn_wmma_f32_16x16x32_f16(false, a, false, b1, (short)0, acc1, false, false);
        acc2 = __builtin_amdgcn_wmma_f32_16x16x32_f16(false, a, false, b2, (short)0, acc2, false, false);
    }

    float* zbase = z + (size_t)(wid * 16) * NCLS;
    #pragma unroll
    for (int r = 0; r < 8; ++r) {
        const size_t row = (size_t)(hf * 8 + r) * NCLS;
        zbase[row + n0]       = acc0[r];
        zbase[row + n0 + 16]  = acc1[r];
        if (n0 < 8) zbase[row + n0 + 32] = acc2[r];
    }
}

// ---------------- scatter layer 2: one wave per edge, 40 feats ----------------
__global__ void k_scatter2(const int* __restrict__ src, const int* __restrict__ dst,
                           const float* __restrict__ dinv,
                           const float* __restrict__ z, float* __restrict__ agg) {
    const int e = (blockIdx.x * blockDim.x + threadIdx.x) >> 5;
    if (e >= NEDGES) return;
    const int lane = threadIdx.x & 31;
    const int s = src[e], d = dst[e];
    const float nrm = dinv[s] * dinv[d];
    const float* zs = z   + (size_t)s * NCLS;
    float*       ad = agg + (size_t)d * NCLS;
    atomAddF(&ad[lane], zs[lane] * nrm);
    if (lane < 8) atomAddF(&ad[lane + 32], zs[lane + 32] * nrm);
}

// ---------------- finalize2: +selfloop, +bias, log_softmax (wave per row) ----------------
__global__ void k_fin2(const float* __restrict__ agg, const float* __restrict__ z,
                       const float* __restrict__ dinv, const float* __restrict__ b2,
                       float* __restrict__ out) {
    const int row = (blockIdx.x * blockDim.x + threadIdx.x) >> 5;
    if (row >= NNODES) return;
    const int lane = threadIdx.x & 31;
    const float di = dinv[row];
    const float sl = di * di;
    const float* a  = agg + (size_t)row * NCLS;
    const float* zz = z   + (size_t)row * NCLS;

    const float v0 = a[lane] + sl * zz[lane] + b2[lane];
    const float v1 = (lane < 8) ? (a[lane + 32] + sl * zz[lane + 32] + b2[lane + 32])
                                : -__builtin_huge_valf();
    float mx = fmaxf(v0, v1);
    #pragma unroll
    for (int o = 16; o > 0; o >>= 1) mx = fmaxf(mx, __shfl_xor(mx, o, 32));
    float s = __expf(v0 - mx) + ((lane < 8) ? __expf(v1 - mx) : 0.0f);
    #pragma unroll
    for (int o = 16; o > 0; o >>= 1) s += __shfl_xor(s, o, 32);
    const float lse = mx + __logf(s);

    float* o = out + (size_t)row * NCLS;
    o[lane] = v0 - lse;
    if (lane < 8) o[lane + 32] = v1 - lse;
}

// ---------------------------------------------------------------------------
extern "C" void kernel_launch(void* const* d_in, const int* in_sizes, int n_in,
                              void* d_out, int out_size, void* d_ws, size_t ws_size,
                              hipStream_t stream) {
    const float* x   = (const float*)d_in[0];
    const int*   ei  = (const int*)  d_in[1];   // [2, E] flat: src then dst
    const float* W1  = (const float*)d_in[2];
    const float* b1  = (const float*)d_in[3];
    const float* W2  = (const float*)d_in[4];
    const float* b2  = (const float*)d_in[5];
    const int*   idx = (const int*)  d_in[6];
    float* out = (float*)d_out;

    const int* src = ei;
    const int* dst = ei + NEDGES;

    // workspace carve-up (all 256B aligned); total ~97 MB
    char* ws = (char*)d_ws;
    size_t off = 0;
    auto take = [&](size_t bytes) { char* p = ws + off; off += (bytes + 255) & ~(size_t)255; return p; };
    float*    deg   = (float*)   take((size_t)NNODES * 4);          // becomes dinv in place
    float*    h     = (float*)   take((size_t)NNODES * HID * 4);
    float*    agg1  = (float*)   take((size_t)NNODES * HID * 4);
    _Float16* hrelu = (_Float16*)take((size_t)NNODES * HID * 2);
    float*    z     = (float*)   take((size_t)NNODES * NCLS * 4);
    float*    agg2  = (float*)   take((size_t)NNODES * NCLS * 4);
    _Float16* w1p   = (_Float16*)take((size_t)(FIN / 32) * 4 * 512 * 2);  // 64 KB
    _Float16* w2p   = (_Float16*)take((size_t)(HID / 32) * 3 * 512 * 2);  //  6 KB
    (void)ws_size;

    // zero atomic accumulators every call (graph-capture-safe)
    hipMemsetAsync(agg1, 0, (size_t)NNODES * HID  * 4, stream);
    hipMemsetAsync(agg2, 0, (size_t)NNODES * NCLS * 4, stream);

    k_fill1<<<(NNODES + 255) / 256, 256, 0, stream>>>(deg);
    k_deg  <<<(NEDGES + 255) / 256, 256, 0, stream>>>(dst, deg);
    k_rsqrt<<<(NNODES + 255) / 256, 256, 0, stream>>>(deg);     // deg -> dinv

    k_packW1<<<(64 * 32 + 255) / 256, 256, 0, stream>>>(W1, w1p);
    k_packW2<<<1, 256, 0, stream>>>(W2, w2p);

    const int gemm_waves  = NNODES / 16;                         // 6250
    const int gemm_blocks = (gemm_waves * 32 + 255) / 256;       // 782
    k_gemm1<<<gemm_blocks, 256, 0, stream>>>(x, w1p, h);

    const int edge_blocks = (NEDGES * 32 + 255) / 256;           // wave per edge
    k_scatter1<<<edge_blocks, 256, 0, stream>>>(src, dst, deg, h, agg1);

    k_fin1<<<(NNODES * HID + 255) / 256, 256, 0, stream>>>(agg1, h, deg, b1, idx, hrelu);

    k_gemm2<<<gemm_blocks, 256, 0, stream>>>(hrelu, w2p, z);

    k_scatter2<<<edge_blocks, 256, 0, stream>>>(src, dst, deg, z, agg2);

    k_fin2<<<(NNODES * 32 + 255) / 256, 256, 0, stream>>>(agg2, z, deg, b2, out);
}
